// MultiHeadDotProductAttention_74964359184636
// MI455X (gfx1250) — matrix operand order repaired and checked by reading, not compiled, not run on hardware
//
#include <hip/hip_runtime.h>
#include <hip/hip_bf16.h>

typedef __bf16 bf16;
typedef __attribute__((ext_vector_type(16))) __bf16 v16bf;
typedef __attribute__((ext_vector_type(8)))  float  v8f;
typedef __attribute__((ext_vector_type(4)))  int    i32x4;

#define WMMA_BF16(a, b, c) \
  __builtin_amdgcn_wmma_f32_16x16x32_bf16(false, (a), false, (b), (short)0, (c), false, false)

// Problem constants
#define BATCH 2
#define QLEN  2048
#define KLEN  2048
#define EMB   2048
#define NH    32
#define HD    64

struct __align__(16) Bf8 { bf16 v[8]; };
struct __align__(8)  Bf4 { bf16 v[4]; };
union  FragU { uint4 q[2]; v16bf v; };

// ----------------- async global->LDS copy (gfx1250 ASYNCcnt path) ----------
#if defined(__has_builtin)
#  if __has_builtin(__builtin_amdgcn_global_load_async_to_lds_b128) && \
      __has_builtin(__builtin_amdgcn_s_wait_asynccnt)
#    define HAVE_ASYNC_LDS 1
#  endif
#endif

static __device__ __forceinline__ void async_copy16(bf16* lds, const bf16* g) {
#ifdef HAVE_ASYNC_LDS
  __builtin_amdgcn_global_load_async_to_lds_b128(
      (__attribute__((address_space(1))) i32x4*)(void*)const_cast<bf16*>(g),
      (__attribute__((address_space(3))) i32x4*)(void*)lds, 0, 0);
#else
  *reinterpret_cast<uint4*>(lds) = *reinterpret_cast<const uint4*>(g);
#endif
}

static __device__ __forceinline__ void wait_async() {
#ifdef HAVE_ASYNC_LDS
  __builtin_amdgcn_s_wait_asynccnt(0);
#endif
}

// ---------------- WMMA fragment loaders (ISA 7.12.2 layouts) ----------------
// A (16x32 bf16): lane L holds row M=L&15; K chunks {kg*8..+7, 16+kg*8..+7}.
static __device__ __forceinline__ v16bf load_a_frag(const bf16* lds, int rowBase, int ld,
                                                    int kBase, int lane) {
  const int row = rowBase + (lane & 15);
  const int kg  = (lane >> 4) & 1;
  const bf16* p = lds + row * ld + kBase + kg * 8;
  FragU u;
  u.q[0] = *reinterpret_cast<const uint4*>(p);
  u.q[1] = *reinterpret_cast<const uint4*>(p + 16);
  return u.v;
}

// B (32x16 bf16) from an N-major tile: B[k][n] = tile[n*ld + k].
static __device__ __forceinline__ v16bf load_bT_frag(const bf16* lds, int nBase, int ld,
                                                     int kBase, int lane) {
  const int n  = nBase + (lane & 15);
  const int kg = (lane >> 4) & 1;
  const bf16* p = lds + n * ld + kBase + kg * 8;
  FragU u;
  u.q[0] = *reinterpret_cast<const uint4*>(p);
  u.q[1] = *reinterpret_cast<const uint4*>(p + 16);
  return u.v;
}

// ---------------------------------------------------------------------------
// Stage 0a: elementwise f32 -> bf16 cast (activations).
// ---------------------------------------------------------------------------
__global__ __launch_bounds__(256) void cast_bf16_kernel(
    const float* __restrict__ in, bf16* __restrict__ out) {
  size_t i = ((size_t)blockIdx.x * 256 + threadIdx.x) * 4;
  float4 f = *reinterpret_cast<const float4*>(in + i);
  Bf4 o; o.v[0] = (bf16)f.x; o.v[1] = (bf16)f.y; o.v[2] = (bf16)f.z; o.v[3] = (bf16)f.w;
  *reinterpret_cast<Bf4*>(out + i) = o;
}

// ---------------------------------------------------------------------------
// Stage 0b: transpose-pack f32 [R][C] -> bf16 [C][R] (weights, packed once).
// ---------------------------------------------------------------------------
__global__ __launch_bounds__(256) void transpose_pack_kernel(
    const float* __restrict__ in, bf16* __restrict__ out, int R, int C) {
  __shared__ float tile[32][33];
  const int tx = threadIdx.x, ty = threadIdx.y;
  const int x = blockIdx.x * 32 + tx;
#pragma unroll
  for (int i = 0; i < 4; ++i) {
    int y = blockIdx.y * 32 + ty + i * 8;
    tile[ty + i * 8][tx] = in[(size_t)y * C + x];
  }
  __syncthreads();
  const int xo = blockIdx.y * 32 + tx;
#pragma unroll
  for (int i = 0; i < 4; ++i) {
    int yo = blockIdx.x * 32 + ty + i * 8;
    out[(size_t)yo * R + xo] = (bf16)tile[tx][ty + i * 8];
  }
}

// ---------------------------------------------------------------------------
// Stage 1: QKV projection GEMM, all bf16, async double-buffered staging.
// C[M=4096][N=2048] = A[M][EMB] x Wt(N-major). Tile 128x128, K-step 32.
// ---------------------------------------------------------------------------
#define ALD 40   // 32 + 8 pad, keeps 16B alignment of fragment chunks
#define NIT (EMB / 32)

template <int VMAJOR>
__global__ __launch_bounds__(256) void proj_gemm_kernel(
    const bf16* __restrict__ A,    // [M][EMB] bf16
    const bf16* __restrict__ Wt,   // [NH*HD][EMB] bf16 (N-major)
    bf16* __restrict__ out, float scale) {
  __shared__ bf16 As[2][128 * ALD];
  __shared__ bf16 Bs[2][128 * ALD];

  const int t    = threadIdx.x;
  const int wave = t >> 5;
  const int lane = t & 31;
  const int rowB = blockIdx.y * 128;
  const int colB = blockIdx.x * 128;

  const int sr0 = (t >> 2);            // staging row (0..63; 2nd copy at +64), 2 Bf8 per thread
  const int sc0 = (t & 3) * 8;         // staging col chunk

  v8f acc[8] = {v8f{}, v8f{}, v8f{}, v8f{}, v8f{}, v8f{}, v8f{}, v8f{}};

  // Prologue: stage tile 0.
#pragma unroll
  for (int i = 0; i < 2; ++i) {
    int r = sr0 + i * 64;
    async_copy16(As[0] + r * ALD + sc0, A  + (size_t)(rowB + r) * EMB + sc0);
    async_copy16(Bs[0] + r * ALD + sc0, Wt + (size_t)(colB + r) * EMB + sc0);
  }

  for (int it = 0; it < NIT; ++it) {
    wait_async();
    __syncthreads();
    if (it + 1 < NIT) {
      const int k1 = (it + 1) * 32;
      bf16* Ad = As[(it + 1) & 1];
      bf16* Bd = Bs[(it + 1) & 1];
#pragma unroll
      for (int i = 0; i < 2; ++i) {
        int r = sr0 + i * 64;
        async_copy16(Ad + r * ALD + sc0, A  + (size_t)(rowB + r) * EMB + k1 + sc0);
        async_copy16(Bd + r * ALD + sc0, Wt + (size_t)(colB + r) * EMB + k1 + sc0);
      }
    }
    const bf16* Ab = As[it & 1];
    const bf16* Bb = Bs[it & 1];
    v16bf a = load_a_frag(Ab, wave * 16, ALD, 0, lane);
#pragma unroll
    for (int nt = 0; nt < 8; ++nt) {
      v16bf b = load_bT_frag(Bb, nt * 16, ALD, 0, lane);
      acc[nt] = WMMA_BF16(a, b, acc[nt]);
    }
  }

  // Scatter to [b,h,s,d] (q/k) or [b,h,d,s] (v). S == 2048 (compile-time).
  const int hi8 = (lane >> 4) ? 8 : 0;
#pragma unroll
  for (int nt = 0; nt < 8; ++nt) {
#pragma unroll
    for (int j = 0; j < 8; ++j) {
      int r = rowB + wave * 16 + j + hi8;
      int c = colB + nt * 16 + (lane & 15);
      int bidx = r >> 11, s = r & 2047;
      int h = c >> 6, d = c & 63;
      size_t idx = VMAJOR ? ((((size_t)bidx * NH + h) * HD + d) * QLEN + s)
                          : ((((size_t)bidx * NH + h) * QLEN + s) * HD + d);
      out[idx] = (bf16)(acc[nt][j] * scale);
    }
  }
}

// ---------------------------------------------------------------------------
// Stage 2: flash-style attention. One (b,h) x 64-query tile per 128-thr block.
// q: [b,h,q,d]; k: [b,h,k,d]; v: [b,h,d,k] (d-major so PV B-frags are contiguous)
// ---------------------------------------------------------------------------
__global__ __launch_bounds__(128) void attn_kernel(
    const bf16* __restrict__ qb, const bf16* __restrict__ kb, const bf16* __restrict__ vb,
    const float* __restrict__ bias, bf16* __restrict__ ctx) {
  __shared__ bf16  Qs[64 * 64];
  __shared__ bf16  Ks[64 * 64];   // [key][d]
  __shared__ bf16  Vt[64 * 64];   // [d][key]
  __shared__ float Ss[64 * 64];   // scores (bias fused in-place)
  __shared__ bf16  Ps[64 * 64];   // probabilities
  __shared__ float rowM[64], rowL[64], rowF[64];

  const int t    = threadIdx.x;
  const int wave = t >> 5;
  const int lane = t & 31;
  const int b    = blockIdx.x >> 5;
  const int h    = blockIdx.x & 31;
  const int q0   = blockIdx.y * 64;
  const int hi8  = (lane >> 4) ? 8 : 0;

  const size_t headBase = ((size_t)b * NH + h) * (size_t)KLEN * HD;

  // Load Q tile (64x64 bf16) via async DMA.
  const bf16* qg = qb + headBase + (size_t)q0 * HD;
#pragma unroll
  for (int i = 0; i < 4; ++i) {
    int idx = t + i * 128;
    int r = idx >> 3, c8 = (idx & 7) * 8;
    async_copy16(Qs + r * 64 + c8, qg + r * HD + c8);
  }
  if (t < 64) { rowM[t] = -INFINITY; rowL[t] = 0.0f; }
  wait_async();
  __syncthreads();

  v8f cacc[4] = {v8f{}, v8f{}, v8f{}, v8f{}};

  for (int kbk = 0; kbk < KLEN / 64; ++kbk) {
    const int kbase = kbk * 64;
    // Stage K tile [key][d] and V tile [d][key] via async DMA.
#pragma unroll
    for (int i = 0; i < 4; ++i) {
      int idx = t + i * 128;
      int r = idx >> 3, c8 = (idx & 7) * 8;
      async_copy16(Ks + r * 64 + c8, kb + headBase + (size_t)(kbase + r) * HD + c8);
      async_copy16(Vt + r * 64 + c8, vb + headBase + (size_t)r * KLEN + kbase + c8);
    }
    wait_async();
    __syncthreads();

    // S = Q x K^T  (16 rows per wave x 64 keys; contract over d=64)
#pragma unroll
    for (int nt = 0; nt < 4; ++nt) {
      v8f sacc = v8f{};
#pragma unroll
      for (int ks = 0; ks < 64; ks += 32) {
        v16bf a  = load_a_frag(Qs, wave * 16, 64, ks, lane);
        v16bf bb = load_bT_frag(Ks, nt * 16, 64, ks, lane);
        sacc = WMMA_BF16(a, bb, sacc);
      }
#pragma unroll
      for (int j = 0; j < 8; ++j)
        Ss[(wave * 16 + j + hi8) * 64 + nt * 16 + (lane & 15)] = sacc[j];
    }
    __syncthreads();

    // Online softmax: thread t<64 owns one query row.
    if (t < 64) {
      const int row = t;
      float* srow = Ss + row * 64;
      bf16*  prow = Ps + row * 64;
      const float* biasRow = bias + ((size_t)b * QLEN + (q0 + row)) * (size_t)KLEN + kbase;
      float mt = -INFINITY;
#pragma unroll
      for (int j4 = 0; j4 < 16; ++j4) {
        float4 bv = *reinterpret_cast<const float4*>(biasRow + j4 * 4);
        float4 sv = *reinterpret_cast<float4*>(srow + j4 * 4);
        sv.x += bv.x; sv.y += bv.y; sv.z += bv.z; sv.w += bv.w;
        *reinterpret_cast<float4*>(srow + j4 * 4) = sv;
        mt = fmaxf(mt, fmaxf(fmaxf(sv.x, sv.y), fmaxf(sv.z, sv.w)));
      }
      float mold = rowM[row];
      float mnew = fmaxf(mold, mt);
      float f = __expf(mold - mnew);
      float l = rowL[row] * f;
#pragma unroll
      for (int j = 0; j < 64; ++j) {
        float p = __expf(srow[j] - mnew);
        l += p;
        prow[j] = (bf16)p;
      }
      rowM[row] = mnew; rowL[row] = l; rowF[row] = f;
    }
    __syncthreads();

    // Rescale ctx accumulators, then ctx += P x V.
    float fj[8];
#pragma unroll
    for (int j = 0; j < 8; ++j) fj[j] = rowF[wave * 16 + j + hi8];
#pragma unroll
    for (int dt = 0; dt < 4; ++dt) {
#pragma unroll
      for (int j = 0; j < 8; ++j) cacc[dt][j] *= fj[j];
#pragma unroll
      for (int ks = 0; ks < 64; ks += 32) {
        v16bf a  = load_a_frag(Ps, wave * 16, 64, ks, lane);
        v16bf bb = load_bT_frag(Vt, dt * 16, 64, ks, lane);  // B[k=key][n=d] = Vt[d][key]
        cacc[dt] = WMMA_BF16(a, bb, cacc[dt]);
      }
    }
    __syncthreads();
  }

  // Normalize and store ctx as bf16 [b, q, h, d].
  float linv[8];
#pragma unroll
  for (int j = 0; j < 8; ++j) linv[j] = 1.0f / rowL[wave * 16 + j + hi8];
#pragma unroll
  for (int dt = 0; dt < 4; ++dt) {
#pragma unroll
    for (int j = 0; j < 8; ++j) {
      int r = q0 + wave * 16 + j + hi8;
      int d = dt * 16 + (lane & 15);
      ctx[(((size_t)b * QLEN + r) * NH + h) * HD + d] = (bf16)(cacc[dt][j] * linv[j]);
    }
  }
}

// ---------------------------------------------------------------------------
// Stage 3: output projection. out(f32)[4096][2048] = ctx(bf16) x wo_t(bf16,N-major)
// ---------------------------------------------------------------------------
__global__ __launch_bounds__(256) void out_gemm_kernel(
    const bf16* __restrict__ A,    // ctx [B*Q][NH*HD] bf16
    const bf16* __restrict__ Wt,   // wo packed [EMB][NH*HD] bf16 (N-major)
    float* __restrict__ out) {     // [B*Q][EMB]
  __shared__ bf16 As[2][128 * ALD];
  __shared__ bf16 Bs[2][128 * ALD];

  const int t    = threadIdx.x;
  const int wave = t >> 5;
  const int lane = t & 31;
  const int rowB = blockIdx.y * 128;
  const int colB = blockIdx.x * 128;
  const int sr0  = (t >> 2);
  const int sc0  = (t & 3) * 8;

  v8f acc[8] = {v8f{}, v8f{}, v8f{}, v8f{}, v8f{}, v8f{}, v8f{}, v8f{}};

#pragma unroll
  for (int i = 0; i < 2; ++i) {
    int r = sr0 + i * 64;
    async_copy16(As[0] + r * ALD + sc0, A  + (size_t)(rowB + r) * (NH * HD) + sc0);
    async_copy16(Bs[0] + r * ALD + sc0, Wt + (size_t)(colB + r) * (NH * HD) + sc0);
  }

  for (int it = 0; it < NIT; ++it) {
    wait_async();
    __syncthreads();
    if (it + 1 < NIT) {
      const int k1 = (it + 1) * 32;
      bf16* Ad = As[(it + 1) & 1];
      bf16* Bd = Bs[(it + 1) & 1];
#pragma unroll
      for (int i = 0; i < 2; ++i) {
        int r = sr0 + i * 64;
        async_copy16(Ad + r * ALD + sc0, A  + (size_t)(rowB + r) * (NH * HD) + k1 + sc0);
        async_copy16(Bd + r * ALD + sc0, Wt + (size_t)(colB + r) * (NH * HD) + k1 + sc0);
      }
    }
    const bf16* Ab = As[it & 1];
    const bf16* Bb = Bs[it & 1];
    v16bf a = load_a_frag(Ab, wave * 16, ALD, 0, lane);
#pragma unroll
    for (int nt = 0; nt < 8; ++nt) {
      v16bf b = load_bT_frag(Bb, nt * 16, ALD, 0, lane);
      acc[nt] = WMMA_BF16(a, b, acc[nt]);
    }
  }

  const int hi8 = (lane >> 4) ? 8 : 0;
#pragma unroll
  for (int nt = 0; nt < 8; ++nt) {
#pragma unroll
    for (int j = 0; j < 8; ++j) {
      int r = rowB + wave * 16 + j + hi8;
      int c = colB + nt * 16 + (lane & 15);
      out[(size_t)r * EMB + c] = acc[nt][j];
    }
  }
}

// ---------------------------------------------------------------------------
extern "C" void kernel_launch(void* const* d_in, const int* in_sizes, int n_in,
                              void* d_out, int out_size, void* d_ws, size_t ws_size,
                              hipStream_t stream) {
  const float* inputs_q  = (const float*)d_in[0];
  const float* inputs_kv = (const float*)d_in[1];
  const float* bias      = (const float*)d_in[2];
  const float* wq        = (const float*)d_in[3];
  const float* wk        = (const float*)d_in[4];
  const float* wv        = (const float*)d_in[5];
  const float* wo        = (const float*)d_in[6];
  float* out             = (float*)d_out;

  const size_t SEG  = (size_t)BATCH * NH * QLEN * HD;  // 8,388,608 bf16
  const size_t WSEG = (size_t)EMB * NH * HD;           // 4,194,304 bf16
  bf16* qws   = (bf16*)d_ws;
  bf16* kws   = qws + SEG;
  bf16* vws   = kws + SEG;
  bf16* ctxws = vws + SEG;
  bf16* wqt   = ctxws + SEG;
  bf16* wkt   = wqt + WSEG;
  bf16* wvt   = wkt + WSEG;
  bf16* wot   = wvt + WSEG;
  bf16* qinb  = wot + WSEG;    // inputs_q  bf16 [B*Q][EMB]
  bf16* kvinb = qinb + SEG;    // inputs_kv bf16 [B*K][EMB]

  // Stage 0: cast activations, pack weights bf16 N-major.
  cast_bf16_kernel<<<SEG / 1024, 256, 0, stream>>>(inputs_q, qinb);
  cast_bf16_kernel<<<SEG / 1024, 256, 0, stream>>>(inputs_kv, kvinb);
  dim3 pg(EMB / 32, EMB / 32), pb(32, 8);
  transpose_pack_kernel<<<pg, pb, 0, stream>>>(wq, wqt, EMB, NH * HD);  // [hd][e]
  transpose_pack_kernel<<<pg, pb, 0, stream>>>(wk, wkt, EMB, NH * HD);
  transpose_pack_kernel<<<pg, pb, 0, stream>>>(wv, wvt, EMB, NH * HD);
  transpose_pack_kernel<<<pg, pb, 0, stream>>>(wo, wot, NH * HD, EMB); // [e][hd]

  // Stage 1: projections (q gets 1/sqrt(head_dim); v stored d-major).
  dim3 gemmGrid((NH * HD) / 128, (BATCH * QLEN) / 128);  // 16 x 32
  proj_gemm_kernel<0><<<gemmGrid, 256, 0, stream>>>(qinb,  wqt, qws, 0.125f);
  proj_gemm_kernel<0><<<gemmGrid, 256, 0, stream>>>(kvinb, wkt, kws, 1.0f);
  proj_gemm_kernel<1><<<gemmGrid, 256, 0, stream>>>(kvinb, wvt, vws, 1.0f);

  // Stage 2: flash attention: grid = (B*H, Q/64)
  attn_kernel<<<dim3(BATCH * NH, QLEN / 64), 128, 0, stream>>>(qws, kws, vws, bias, ctxws);

  // Stage 3: output projection.
  out_gemm_kernel<<<dim3(EMB / 128, (BATCH * QLEN) / 128), 256, 0, stream>>>(ctxws, wot, out);
}
